// LongLlamaAttention_2164663517481
// MI455X (gfx1250) — matrix operand order repaired and checked by reading, not compiled, not run on hardware
//
#include <hip/hip_runtime.h>
#include <hip/hip_bf16.h>

// ---------------------------------------------------------------------------
// LongLlama attention block for gfx1250 (MI455X), bf16 WMMA pipeline.
//   B=1, S=4096, HID=2048, H=16, D=128
// Pipeline: fp32->bf16, QKV GEMMs (WMMA + async-to-LDS double buffering),
//           RoPE, V pre-transpose, flash attention (WMMA scores + WMMA
//           row-sums + DPP row-max), output GEMM (WMMA).
// ---------------------------------------------------------------------------

typedef __bf16 bf16;
typedef __attribute__((ext_vector_type(16))) __bf16 bf16x16;
typedef __attribute__((ext_vector_type(8)))  __bf16 bf16x8;
typedef __attribute__((ext_vector_type(8)))  float  f32x8;
typedef __attribute__((ext_vector_type(4)))  int    v4i;

#define S_LEN 4096
#define HID   2048
#define NHEAD 16
#define DHEAD 128

static __device__ __forceinline__ bf16x16 cat8(bf16x8 a, bf16x8 b) {
    return __builtin_shufflevector(a, b, 0,1,2,3,4,5,6,7,8,9,10,11,12,13,14,15);
}

// ---- async global->LDS copy (16B), tracked by ASYNCcnt ---------------------
static __device__ __forceinline__ void async_ld16(const bf16* g, bf16* l) {
#if __has_builtin(__builtin_amdgcn_global_load_async_to_lds_b128)
    __builtin_amdgcn_global_load_async_to_lds_b128(
        (__attribute__((address_space(1))) v4i*)g,
        (__attribute__((address_space(3))) v4i*)l, 0, 0);
#else
    unsigned laddr = (unsigned)(size_t)(__attribute__((address_space(3))) void*)l;
    asm volatile("global_load_async_to_lds_b128 %0, %1, off"
                 :: "v"(laddr), "v"(g) : "memory");
#endif
}

template <int N>
static __device__ __forceinline__ void wait_async() {
#if __has_builtin(__builtin_amdgcn_s_wait_asynccnt)
    __builtin_amdgcn_s_wait_asynccnt(N);
#else
    asm volatile("s_wait_asynccnt %0" :: "i"(N) : "memory");
#endif
}

// DPP butterfly step: max with a permuted copy (stays within 16-lane row).
template <int CTRL>
static __device__ __forceinline__ float fmax_dpp(float v) {
    int t = __builtin_amdgcn_update_dpp(0, __float_as_int(v), CTRL, 0xF, 0xF, true);
    return fmaxf(v, __int_as_float(t));
}
// Max across each group of 16 lanes (lanes 0-15 / 16-31 independent).
static __device__ __forceinline__ float rowmax16(float v) {
    v = fmax_dpp<0xB1>(v);   // quad_perm(1,0,3,2): xor 1
    v = fmax_dpp<0x4E>(v);   // quad_perm(2,3,0,1): xor 2
    v = fmax_dpp<0x141>(v);  // row_half_mirror:    cross 4-group
    v = fmax_dpp<0x140>(v);  // row_mirror:         cross 8-group
    return v;
}

// --------------------------- fp32 -> bf16 ----------------------------------
__global__ void cvt_f32_bf16_kernel(const float* __restrict__ in,
                                    bf16* __restrict__ out, int n) {
    for (int i = blockIdx.x * blockDim.x + threadIdx.x; i < n;
         i += gridDim.x * blockDim.x)
        out[i] = (bf16)in[i];
}

// --------------------------- min(position_ids) -----------------------------
__global__ void minpos_kernel(const int* __restrict__ pos, int n,
                              int* __restrict__ out) {
    __shared__ int sm[256];
    int t = threadIdx.x;
    int v = 0x7fffffff;
    for (int i = t; i < n; i += 256) v = min(v, pos[i]);
    sm[t] = v;
    __syncthreads();
    for (int s = 128; s > 0; s >>= 1) {
        if (t < s) sm[t] = min(sm[t], sm[t + s]);
        __syncthreads();
    }
    if (t == 0) out[0] = sm[0];
}

// --------------------------- RoPE (in-place, bf16) --------------------------
__global__ void rope_kernel(bf16* __restrict__ buf, const int* __restrict__ pos,
                            const int* __restrict__ minpos, float scale) {
    int tid = blockIdx.x * blockDim.x + threadIdx.x;
    if (tid >= S_LEN * NHEAD * 64) return;
    int d = tid & 63;
    int h = (tid >> 6) & (NHEAD - 1);
    int s = tid >> 10;
    size_t base = (size_t)s * HID + h * DHEAD + d;
    float x1 = (float)buf[base];
    float x2 = (float)buf[base + 64];
    float p = (float)(pos[s] - minpos[0]);
    float ang = p * __powf(10000.0f, -(float)d * (1.0f / 64.0f));
    float sn, cs;
    __sincosf(ang, &sn, &cs);
    buf[base]      = (bf16)((x1 * cs - x2 * sn) * scale);
    buf[base + 64] = (bf16)((x2 * cs + x1 * sn) * scale);
}

// --------------------------- V transpose: [S,HID] -> [HID,S] ----------------
__global__ __launch_bounds__(256)
void transpose_kernel(const bf16* __restrict__ in, bf16* __restrict__ out) {
    __shared__ bf16 tile[32][33];
    int tx = threadIdx.x;        // 0..31
    int ty = threadIdx.y;        // 0..7
    int s0 = blockIdx.x * 32;    // row tile (seq)
    int c0 = blockIdx.y * 32;    // col tile (hid)
#pragma unroll
    for (int i = 0; i < 4; ++i)
        tile[ty + 8 * i][tx] = in[(size_t)(s0 + ty + 8 * i) * HID + c0 + tx];
    __syncthreads();
#pragma unroll
    for (int i = 0; i < 4; ++i)
        out[(size_t)(c0 + ty + 8 * i) * S_LEN + s0 + tx] = tile[tx][ty + 8 * i];
}

// --------------------------- WMMA GEMM: C = A * W^T -------------------------
// Block-cooperative: 256 threads compute a 128x128 C block; 128x32 A/W panels
// are staged into LDS with async-to-LDS loads, double-buffered on ASYNCcnt.
// Each wave computes a 32x64 subtile (8 WMMA per K-step from LDS fragments).
#define LSTR 40  // LDS panel row stride (elems): 16B-aligned, bank-spread

template <bool OUTF32>
__global__ __launch_bounds__(256)
void gemm_wmma_kernel(const bf16* __restrict__ A, const bf16* __restrict__ W,
                      void* __restrict__ Cout, int M, int N, int K) {
    __shared__ bf16 As[2][128 * LSTR];
    __shared__ bf16 Ws[2][128 * LSTR];

    int nblocks = N >> 7;
    int m0b = (int)(blockIdx.x / nblocks) << 7;
    int n0b = (int)(blockIdx.x % nblocks) << 7;
    int t = threadIdx.x;
    int w = t >> 5;
    int mw = (w & 3) << 5;   // wave's m-subtile: 0,32,64,96
    int nw = (w >> 2) << 6;  // wave's n-subtile: 0,64
    int lane = t & 31, col = lane & 15, hi = lane >> 4;

    // Staging coords: 512 16B-chunks per 128x32 panel, 2 per thread.
    int r0 = t >> 1,  q0 = (t & 1) << 4;          // rows 0..127, elem 0/16
    int r1 = r0, q1 = q0 + 8;                     // second 16B of same 32B

    auto stage = [&](int buf, int k0) {
        async_ld16(A + (size_t)(m0b + r0) * K + k0 + q0, &As[buf][r0 * LSTR + q0]);
        async_ld16(A + (size_t)(m0b + r1) * K + k0 + q1, &As[buf][r1 * LSTR + q1]);
        async_ld16(W + (size_t)(n0b + r0) * K + k0 + q0, &Ws[buf][r0 * LSTR + q0]);
        async_ld16(W + (size_t)(n0b + r1) * K + k0 + q1, &Ws[buf][r1 * LSTR + q1]);
    };

    f32x8 acc[2][4];
#pragma unroll
    for (int r = 0; r < 2; ++r)
#pragma unroll
        for (int s = 0; s < 4; ++s) acc[r][s] = 0.0f;

    stage(0, 0);
    for (int k0 = 0, it = 0; k0 < K; k0 += 32, ++it) {
        int buf = it & 1;
        if (k0 + 32 < K) {
            stage(buf ^ 1, k0 + 32);   // prefetch next panel
            wait_async<4>();           // this panel's 4 loads complete (in-order)
        } else {
            wait_async<0>();
        }
        __syncthreads();

        const bf16* ar0 = &As[buf][(mw + col) * LSTR];
        const bf16* ar1 = &As[buf][(mw + 16 + col) * LSTR];
        bf16x16 a0 = cat8(*(const bf16x8*)(ar0 + 16 * hi),
                          *(const bf16x8*)(ar0 + 8 + 16 * hi));
        bf16x16 a1 = cat8(*(const bf16x8*)(ar1 + 16 * hi),
                          *(const bf16x8*)(ar1 + 8 + 16 * hi));
#pragma unroll
        for (int s = 0; s < 4; ++s) {
            const bf16* wr = &Ws[buf][(nw + 16 * s + col) * LSTR + 16 * hi];
            bf16x16 b = cat8(*(const bf16x8*)(wr), *(const bf16x8*)(wr + 8));
            acc[0][s] = __builtin_amdgcn_wmma_f32_16x16x32_bf16(
                false, a0, false, b, (short)0, acc[0][s], false, false);
            acc[1][s] = __builtin_amdgcn_wmma_f32_16x16x32_bf16(
                false, a1, false, b, (short)0, acc[1][s], false, false);
        }
        __syncthreads();  // all waves done with this panel before overwrite
    }

#pragma unroll
    for (int r = 0; r < 2; ++r) {
        int rbase = m0b + mw + 16 * r + 8 * hi;  // C layout: row = j + 8*hi
#pragma unroll
        for (int s = 0; s < 4; ++s) {
            int c0 = n0b + nw + 16 * s + col;
#pragma unroll
            for (int j = 0; j < 8; ++j) {
                float v = acc[r][s][j];
                size_t idx = (size_t)(rbase + j) * N + c0;
                if (OUTF32) ((float*)Cout)[idx] = v;
                else        ((bf16*)Cout)[idx]  = (bf16)v;
            }
        }
    }
}

// --------------------------- Flash attention --------------------------------
// One wave per (head, 16-row Q tile), 64 keys per iteration. Online softmax:
//   - row max: DPP butterfly (no LDS)
//   - row sum: P x ONES via WMMA (folded into accumulator rescale)
// Causal mask analytic, applied only on diagonal tiles. Q pre-scaled 1/sqrt(D).
// Q/K layout: [S,HID] bf16. Vt layout: [HID,S] bf16 -> contiguous B-fragments.
#define PP 72  // padded row stride of P tile in LDS (16B-aligned frag reads)

__global__ __launch_bounds__(128)
void flash_attn_kernel(const bf16* __restrict__ Q, const bf16* __restrict__ K,
                       const bf16* __restrict__ Vt, bf16* __restrict__ O) {
    __shared__ bf16 ldsP[4][16 * PP];  // P tile: [qrow][key 0..63], per wave

    int wib  = threadIdx.x >> 5;
    int wid  = blockIdx.x * 4 + wib;
    int head = wid >> 8;          // 256 q-tiles per head
    int qt   = wid & 255;
    int q0   = qt << 4;
    int lane = threadIdx.x & 31;
    int col  = lane & 15;
    int hi   = lane >> 4;

    bf16* pp = ldsP[wib];

    // Preload Q A-fragments (4 chunks of K-dim 32 covering D=128).
    const bf16* Qr = Q + (size_t)(q0 + col) * HID + head * DHEAD;
    bf16x16 aq[4];
#pragma unroll
    for (int f = 0; f < 4; ++f)
        aq[f] = cat8(*(const bf16x8*)(Qr + 32 * f + 16 * hi),
                     *(const bf16x8*)(Qr + 32 * f + 8 + 16 * hi));

    const bf16* Kh  = K + head * DHEAD + 16 * hi;
    const bf16* Vth = Vt + (size_t)head * DHEAD * S_LEN;

    // All-ones B fragment: lsum += P x ONES gives per-row sums of P.
    bf16x16 ones;
#pragma unroll
    for (int i = 0; i < 16; ++i) ones[i] = (bf16)1.0f;

    float m[8];
    f32x8 o[8], lsum;
#pragma unroll
    for (int j = 0; j < 8; ++j) m[j] = -3.0e38f;
#pragma unroll
    for (int t = 0; t < 8; ++t) o[t] = 0.0f;
    lsum = 0.0f;

    int ktmax = (q0 + 15) >> 6;  // causal: keys 0 .. q0+15, 64 keys/tile
    for (int kt = 0; kt <= ktmax; ++kt) {
        int k0 = kt << 6;

        // ---- scores: S[16q x 64k] = Q x K^T (four 16x16 subtiles) ----
        f32x8 sc[4];
#pragma unroll
        for (int n = 0; n < 4; ++n) {
            sc[n] = 0.0f;
            const bf16* Kr = Kh + (size_t)(k0 + 16 * n + col) * HID;
#pragma unroll
            for (int f = 0; f < 4; ++f) {
                bf16x16 b = *(const bf16x16*)(Kr + 32 * f);
                sc[n] = __builtin_amdgcn_wmma_f32_16x16x32_bf16(
                    false, aq[f], false, b, (short)0, sc[n], false, false);
            }
        }

        // ---- causal mask (diagonal tiles only, wave-uniform branch) ----
        if (k0 + 63 > q0) {
#pragma unroll
            for (int n = 0; n < 4; ++n) {
                int kk = k0 + 16 * n + col;
#pragma unroll
                for (int j = 0; j < 8; ++j)
                    if (kk > q0 + j + 8 * hi) sc[n][j] = -3.0e38f;
            }
        }

        // ---- online softmax: DPP row-max, exp, P -> LDS ----
        float corr[8];
#pragma unroll
        for (int j = 0; j < 8; ++j) {
            float v = fmaxf(fmaxf(sc[0][j], sc[1][j]),
                            fmaxf(sc[2][j], sc[3][j]));
            v = rowmax16(v);
            float mn = fmaxf(m[j], v);
            corr[j] = __expf(m[j] - mn);
            m[j] = mn;
            int prow = (j + 8 * hi) * PP;
#pragma unroll
            for (int n = 0; n < 4; ++n) {
                float pn = __expf(sc[n][j] - mn);
                pp[prow + 16 * n + col] = (bf16)pn;
            }
        }
#pragma unroll
        for (int t = 0; t < 8; ++t)
#pragma unroll
            for (int j = 0; j < 8; ++j) o[t][j] *= corr[j];
#pragma unroll
        for (int j = 0; j < 8; ++j) lsum[j] *= corr[j];

        // ---- P fragments (A-layout) for keys [k0,k0+32) and [k0+32,k0+64) ----
        const bf16* pr = pp + col * PP;
        bf16x16 ap0 = cat8(*(const bf16x8*)(pr + 16 * hi),
                           *(const bf16x8*)(pr + 8 + 16 * hi));
        bf16x16 ap1 = cat8(*(const bf16x8*)(pr + 32 + 16 * hi),
                           *(const bf16x8*)(pr + 40 + 16 * hi));

        // ---- row sums: lsum += P x ONES ----
        lsum = __builtin_amdgcn_wmma_f32_16x16x32_bf16(
            false, ap0, false, ones, (short)0, lsum, false, false);
        lsum = __builtin_amdgcn_wmma_f32_16x16x32_bf16(
            false, ap1, false, ones, (short)0, lsum, false, false);

        // ---- O += P x V  (B-fragments contiguous in Vt) ----
#pragma unroll
        for (int t = 0; t < 8; ++t) {
            const bf16* vb = Vth + (size_t)(16 * t + col) * S_LEN + k0 + 16 * hi;
            bf16x16 bv0 = *(const bf16x16*)(vb);
            bf16x16 bv1 = *(const bf16x16*)(vb + 32);
            o[t] = __builtin_amdgcn_wmma_f32_16x16x32_bf16(
                false, ap0, false, bv0, (short)0, o[t], false, false);
            o[t] = __builtin_amdgcn_wmma_f32_16x16x32_bf16(
                false, ap1, false, bv1, (short)0, o[t], false, false);
        }
    }

    // ---- epilogue: O / l, write bf16 (lsum columns all hold the row sum) ----
    float invl[8];
#pragma unroll
    for (int j = 0; j < 8; ++j) invl[j] = 1.0f / lsum[j];
#pragma unroll
    for (int t = 0; t < 8; ++t) {
        int dcol = head * DHEAD + 16 * t + col;
#pragma unroll
        for (int j = 0; j < 8; ++j) {
            size_t idx = (size_t)(q0 + j + 8 * hi) * HID + dcol;
            O[idx] = (bf16)(o[t][j] * invl[j]);
        }
    }
}

// ---------------------------------------------------------------------------
extern "C" void kernel_launch(void* const* d_in, const int* in_sizes, int n_in,
                              void* d_out, int out_size, void* d_ws, size_t ws_size,
                              hipStream_t stream) {
    (void)in_sizes; (void)n_in; (void)out_size; (void)ws_size;
    const float* hidden = (const float*)d_in[0];
    // d_in[1] = attention_mask: exact causal mask, applied analytically instead.
    const int* pos = (const int*)d_in[2];
    const float* Wq = (const float*)d_in[3];
    const float* Wk = (const float*)d_in[4];
    const float* Wv = (const float*)d_in[5];
    const float* Wo = (const float*)d_in[6];
    float* out = (float*)d_out;

    const size_t SZX = (size_t)S_LEN * HID;  // 8.39M elems
    const size_t SZW = (size_t)HID * HID;    // 4.19M elems

    char* p = (char*)d_ws;
    bf16* Xb  = (bf16*)p; p += SZX * 2;
    bf16* Wqb = (bf16*)p; p += SZW * 2;
    bf16* Wkb = (bf16*)p; p += SZW * 2;
    bf16* Wvb = (bf16*)p; p += SZW * 2;
    bf16* Wob = (bf16*)p; p += SZW * 2;
    bf16* Qb  = (bf16*)p; p += SZX * 2;
    bf16* Kb  = (bf16*)p; p += SZX * 2;
    bf16* Vb  = (bf16*)p; p += SZX * 2;
    bf16* Vtb = (bf16*)p; p += SZX * 2;
    bf16* Ob  = (bf16*)p; p += SZX * 2;
    int* minp = (int*)p;

    // 1. Convert to bf16.
    cvt_f32_bf16_kernel<<<2048, 256, 0, stream>>>(hidden, Xb, (int)SZX);
    cvt_f32_bf16_kernel<<<2048, 256, 0, stream>>>(Wq, Wqb, (int)SZW);
    cvt_f32_bf16_kernel<<<2048, 256, 0, stream>>>(Wk, Wkb, (int)SZW);
    cvt_f32_bf16_kernel<<<2048, 256, 0, stream>>>(Wv, Wvb, (int)SZW);
    cvt_f32_bf16_kernel<<<2048, 256, 0, stream>>>(Wo, Wob, (int)SZW);

    // 2. min(position_ids).
    minpos_kernel<<<1, 256, 0, stream>>>(pos, S_LEN, minp);

    // 3. QKV projections: [4096,2048] x [2048,2048]^T, 128x128 block/WG.
    int gemm_blocks = (S_LEN / 128) * (HID / 128);  // 512
    gemm_wmma_kernel<false><<<gemm_blocks, 256, 0, stream>>>(Xb, Wqb, Qb, S_LEN, HID, HID);
    gemm_wmma_kernel<false><<<gemm_blocks, 256, 0, stream>>>(Xb, Wkb, Kb, S_LEN, HID, HID);
    gemm_wmma_kernel<false><<<gemm_blocks, 256, 0, stream>>>(Xb, Wvb, Vb, S_LEN, HID, HID);

    // 4. RoPE (scale = 1/sqrt(128) folded into Q).
    int rope_threads = S_LEN * NHEAD * 64;
    rope_kernel<<<(rope_threads + 255) / 256, 256, 0, stream>>>(
        Qb, pos, minp, 0.08838834764831845f);
    rope_kernel<<<(rope_threads + 255) / 256, 256, 0, stream>>>(
        Kb, pos, minp, 1.0f);

    // 5. Pre-transpose V: [S,HID] -> [HID,S] so attention reads are contiguous.
    transpose_kernel<<<dim3(S_LEN / 32, HID / 32), dim3(32, 8), 0, stream>>>(Vb, Vtb);

    // 6. Flash attention: 16 heads x 256 q-tiles = 4096 waves, 4 waves/block.
    flash_attn_kernel<<<1024, 128, 0, stream>>>(Qb, Kb, Vtb, Ob);

    // 7. Output projection -> fp32 d_out.
    gemm_wmma_kernel<true><<<gemm_blocks, 256, 0, stream>>>(Ob, Wob, (void*)out,
                                                            S_LEN, HID, HID);
}
